// DecoderRNN_Attn_22179211116824
// MI455X (gfx1250) — compile-verified, matrix operand order checked by reading, not run on hardware
//
#include <hip/hip_runtime.h>
#include <hip/hip_bf16.h>

// ---------------------------------------------------------------------------
// DecoderRNN with Luong "general" attention, GRU cell, teacher forcing.
// GEMMs: v_wmma_f32_16x16x32_bf16, A tile staged to LDS via async-to-LDS copy
// (ASYNCcnt), 4 N-tiles per wave; fp32 accumulate. Softmax/GRU math in fp32.
// Streaming logp output uses non-temporal stores to protect W_out residency
// in L2 (33MB bf16 weights vs 156MB of streamed output per call).
// ---------------------------------------------------------------------------

#define B_ 64
#define S_ 64
#define T_ 20
#define H_ 512
#define E_ 256
#define V_ 32000
#define TSTEPS (T_ - 1)   // 19 decode steps
#define KMAX_ 1024        // largest GEMM K (W_concat: 2H)

typedef __attribute__((ext_vector_type(16))) __bf16 v16bf;
typedef __attribute__((ext_vector_type(8)))  __bf16 v8bf;
typedef __attribute__((ext_vector_type(8)))  float  v8f;

// exact pointee type expected by the async-to-LDS builtin (from diagnostics):
typedef int v4i_vs __attribute__((vector_size(16)));
typedef __attribute__((address_space(1))) v4i_vs* gptr_b128;
typedef __attribute__((address_space(3))) v4i_vs* lptr_b128;

__device__ inline __bf16 f2b(float x) { return (__bf16)x; }

#if __has_builtin(__builtin_amdgcn_global_load_async_to_lds_b128) && \
    __has_builtin(__builtin_amdgcn_s_wait_asynccnt)
#define USE_ASYNC_LDS 1
#else
#define USE_ASYNC_LDS 0
#endif

// ---------------------------------------------------------------------------
// WMMA GEMM: C[m][n] = epi( sum_k A[m][k]*B[n][k] + bias[n] )
//   A: (M x K) bf16 row-major with lda == K (contiguous tiles)
//   B: (N x K) bf16 row-major (X @ W^T shape)
//   grid = (ceil(tilesN/32), tilesM); block = 256 (8 waves); each wave: 4 tiles
//   A tile (16 x K) staged to LDS once per block (async-to-LDS when available)
// ---------------------------------------------------------------------------
template <int EPI>   // 0: linear (+bias), 1: tanh(+bias)
__global__ void wmma_gemm_kernel(const __bf16* __restrict__ A,
                                 const __bf16* __restrict__ B, int ldb,
                                 const float* __restrict__ bias,
                                 float* __restrict__ C, int ldc,
                                 __bf16* __restrict__ Cbf,
                                 int tilesN, int K) {
  __shared__ __bf16 Alds[16 * KMAX_];
  const int tid  = threadIdx.x;
  const int lane = tid & 31;
  const int wv   = tid >> 5;                       // wave in block: 0..7
  const long m0  = (long)blockIdx.y * 16;

  // ---- cooperative stage of the contiguous 16xK A tile into LDS ----
  const __bf16* gsrc = A + m0 * (long)K;           // lda == K
  const int chunks = (16 * K) >> 3;                // 16B chunks
#if USE_ASYNC_LDS
  for (int c = tid; c < chunks; c += 256) {
    __builtin_amdgcn_global_load_async_to_lds_b128(
        (gptr_b128)(uintptr_t)(gsrc + ((long)c << 3)),
        (lptr_b128)(unsigned)(uintptr_t)(Alds + ((long)c << 3)),
        0, 0);
  }
  __builtin_amdgcn_s_wait_asynccnt(0);
#else
  for (int c = tid; c < chunks; c += 256) {
    *(v8bf*)(Alds + ((long)c << 3)) = *(const v8bf*)(gsrc + ((long)c << 3));
  }
#endif
  __syncthreads();

  // ---- compute 4 consecutive 16x16 tiles per wave ----
  const int tn0  = blockIdx.x * 32 + wv * 4;
  const int r    = lane & 15;                      // A row / B column in tile
  const int half = lane >> 4;                      // K-half selector (ISA 7.12.2)
  const __bf16* als = Alds + (long)r * K;

  const __bf16* brow[4];
#pragma unroll
  for (int mt = 0; mt < 4; ++mt) {
    int tn = tn0 + mt;
    tn = (tn < tilesN) ? tn : (tilesN - 1);        // clamp loads, mask stores later
    brow[mt] = B + ((long)tn * 16 + r) * (long)ldb;
  }

  v8f acc[4];
#pragma unroll
  for (int mt = 0; mt < 4; ++mt) acc[mt] = (v8f){0.f,0.f,0.f,0.f,0.f,0.f,0.f,0.f};

  for (int k0 = 0; k0 < K; k0 += 32) {
    __builtin_prefetch(brow[3] + k0 + 32, 0, 0);   // speculative, never faults
    v8bf alo = *(const v8bf*)(als + k0 + (half << 3));
    v8bf ahi = *(const v8bf*)(als + k0 + 16 + (half << 3));
    v16bf a = __builtin_shufflevector(alo, ahi, 0,1,2,3,4,5,6,7,8,9,10,11,12,13,14,15);
#pragma unroll
    for (int mt = 0; mt < 4; ++mt) {
      v8bf blo = *(const v8bf*)(brow[mt] + k0 + (half << 3));
      v8bf bhi = *(const v8bf*)(brow[mt] + k0 + 16 + (half << 3));
      v16bf b = __builtin_shufflevector(blo, bhi, 0,1,2,3,4,5,6,7,8,9,10,11,12,13,14,15);
      acc[mt] = __builtin_amdgcn_wmma_f32_16x16x32_bf16(
          false, a, false, b, (short)0, acc[mt], false, false);
    }
  }

  // ---- epilogue: C/D layout m = rr + 8*half, n = lane%16 ----
#pragma unroll
  for (int mt = 0; mt < 4; ++mt) {
    if (tn0 + mt >= tilesN) break;                 // wave-uniform
    long n = (long)(tn0 + mt) * 16 + (lane & 15);
    float bv = bias ? bias[n] : 0.f;
#pragma unroll
    for (int rr = 0; rr < 8; ++rr) {
      long m = m0 + rr + (half << 3);
      float val = acc[mt][rr] + bv;
      if (EPI == 1) val = tanhf(val);
      C[m * (long)ldc + n] = val;
      if (Cbf) Cbf[m * (long)ldc + n] = f2b(val);
    }
  }
}

// ---------------------------------------------------------------------------
// Setup / elementwise kernels
// ---------------------------------------------------------------------------
__global__ void f2bf_kernel(const float* __restrict__ src, __bf16* __restrict__ dst,
                            long n) {
  long i = (long)blockIdx.x * blockDim.x + threadIdx.x;
  if (i < n) dst[i] = f2b(src[i]);
}

// enc_outputs (S,B,H) fp32 -> enc_bsh (B*S, H) bf16  (row = b*S + s)
__global__ void enc_transpose_kernel(const float* __restrict__ enc,
                                     __bf16* __restrict__ dst) {
  long i = (long)blockIdx.x * blockDim.x + threadIdx.x;
  if (i >= (long)B_ * S_ * H_) return;
  int j = (int)(i & (H_ - 1));
  long row = i >> 9;                      // H_=512
  int s = (int)(row & (S_ - 1));
  int b = (int)(row >> 6);                // S_=64
  dst[i] = f2b(enc[((long)s * B_ + b) * H_ + j]);
}

// h0 = enc_final[0]; attn0 = 0
__global__ void init_state_kernel(const float* __restrict__ enc_final,
                                  float* __restrict__ h, __bf16* __restrict__ h_bf,
                                  __bf16* __restrict__ attn_bf) {
  int i = blockIdx.x * blockDim.x + threadIdx.x;
  if (i >= B_ * H_) return;
  float v = enc_final[i];
  h[i] = v;
  h_bf[i] = f2b(v);
  attn_bf[i] = f2b(0.f);
}

// x = [ emb[tgt[b][t]], attn_h ]  -> bf16 (B x (E+H))
__global__ void build_x_kernel(const int* __restrict__ tgt,
                               const float* __restrict__ emb,
                               const __bf16* __restrict__ attn_bf,
                               __bf16* __restrict__ x_bf, int t) {
  int i = blockIdx.x * blockDim.x + threadIdx.x;
  if (i >= B_ * (E_ + H_)) return;
  int b = i / (E_ + H_);
  int j = i % (E_ + H_);
  if (j < E_) {
    int tok = tgt[b * T_ + t];
    x_bf[i] = f2b(emb[(long)tok * E_ + j]);
  } else {
    x_bf[i] = attn_bf[b * H_ + (j - E_)];
  }
}

// GRU gate combine: gi/gh are (B x 3H) fp32 (bias already added in GEMM)
__global__ void gru_combine_kernel(const float* __restrict__ gi,
                                   const float* __restrict__ gh,
                                   float* __restrict__ h, __bf16* __restrict__ h_bf) {
  int i = blockIdx.x * blockDim.x + threadIdx.x;
  if (i >= B_ * H_) return;
  int b = i >> 9;
  int j = i & (H_ - 1);
  const float* gib = gi + (long)b * 3 * H_;
  const float* ghb = gh + (long)b * 3 * H_;
  float r = 1.f / (1.f + expf(-(gib[j] + ghb[j])));
  float z = 1.f / (1.f + expf(-(gib[H_ + j] + ghb[H_ + j])));
  float n = tanhf(gib[2 * H_ + j] + r * ghb[2 * H_ + j]);
  float hv = (1.f - z) * n + z * h[i];
  h[i] = hv;
  h_bf[i] = f2b(hv);
}

// Attention: score/softmax/context; writes cat_bf = [h | ctx] (B x 2H) bf16.
__global__ void attn_kernel(const float* __restrict__ h,
                            const __bf16* __restrict__ h_bf,
                            const float* __restrict__ enc_proj,    // (B*S, H) fp32
                            const float* __restrict__ enc_outputs, // (S,B,H) fp32
                            const int* __restrict__ len_src,
                            __bf16* __restrict__ cat_bf) {
  int b = blockIdx.x;
  int t = threadIdx.x;                 // 256 threads
  __shared__ float sc[S_];
  __shared__ float aw[S_];
  int s  = t >> 2;                     // 4 threads per source position
  int l4 = t & 3;
  const float* hp = h + (long)b * H_;
  const float* ep = enc_proj + ((long)b * S_ + s) * H_;
  float p = 0.f;
  for (int k = l4; k < H_; k += 4) p += hp[k] * ep[k];
  p += __shfl_xor(p, 1, 32);
  p += __shfl_xor(p, 2, 32);
  if (l4 == 0) sc[s] = (s < len_src[b]) ? p : -1e9f;
  __syncthreads();
  if (t == 0) {
    float mx = sc[0];
    for (int i = 1; i < S_; ++i) mx = fmaxf(mx, sc[i]);
    float sum = 0.f;
    for (int i = 0; i < S_; ++i) { float e = expf(sc[i] - mx); aw[i] = e; sum += e; }
    float inv = 1.f / sum;
    for (int i = 0; i < S_; ++i) aw[i] *= inv;
  }
  __syncthreads();
  for (int j = t; j < H_; j += 256) {
    float c = 0.f;
    for (int i = 0; i < S_; ++i)
      c += aw[i] * enc_outputs[((long)i * B_ + b) * H_ + j];
    cat_bf[(long)b * 2 * H_ + H_ + j] = f2b(c);
    cat_bf[(long)b * 2 * H_ + j]      = h_bf[(long)b * H_ + j];
  }
}

// Per-row logits stats: max, log(sum exp(x-max)), argmax (first-index ties).
__global__ void logits_stats_kernel(const float* __restrict__ logits,
                                    float* __restrict__ stats,
                                    float* __restrict__ words_out) {
  int b = blockIdx.x;
  int t = threadIdx.x;                 // 256 threads
  __shared__ float smax[256];
  __shared__ int   sarg[256];
  __shared__ float ssum[256];
  const float* row = logits + (long)b * V_;
  float mx = -3.4e38f; int am = 0x7fffffff;
  for (int v = t; v < V_; v += 256) {
    float x = row[v];
    if (x > mx) { mx = x; am = v; }
  }
  smax[t] = mx; sarg[t] = am;
  __syncthreads();
  for (int w = 128; w > 0; w >>= 1) {
    if (t < w) {
      if (smax[t + w] > smax[t] ||
          (smax[t + w] == smax[t] && sarg[t + w] < sarg[t])) {
        smax[t] = smax[t + w]; sarg[t] = sarg[t + w];
      }
    }
    __syncthreads();
  }
  float rmx = smax[0];
  float sum = 0.f;
  for (int v = t; v < V_; v += 256) sum += expf(row[v] - rmx);
  ssum[t] = sum;
  __syncthreads();
  for (int w = 128; w > 0; w >>= 1) {
    if (t < w) ssum[t] += ssum[t + w];
    __syncthreads();
  }
  if (t == 0) {
    stats[b * 2]     = rmx;
    stats[b * 2 + 1] = logf(ssum[0]);
    words_out[b] = (float)sarg[0];
  }
}

// logp = logits - max - log(sumexp); streamed with non-temporal stores so the
// 156MB of output writes don't evict W_out_bf16 (33MB, reused 19x) from L2.
__global__ void write_logp_kernel(const float* __restrict__ logits,
                                  const float* __restrict__ stats,
                                  float* __restrict__ out_lp) {
  long i = (long)blockIdx.x * blockDim.x + threadIdx.x;
  if (i >= (long)B_ * V_) return;
  int b = (int)(i / V_);
  float v = __builtin_nontemporal_load(&logits[i]) - stats[b * 2] - stats[b * 2 + 1];
  __builtin_nontemporal_store(v, &out_lp[i]);
}

// ---------------------------------------------------------------------------
// Launch
// ---------------------------------------------------------------------------
static inline int cdiv(long n, int d) { return (int)((n + d - 1) / d); }

extern "C" void kernel_launch(void* const* d_in, const int* in_sizes, int n_in,
                              void* d_out, int out_size, void* d_ws, size_t ws_size,
                              hipStream_t stream) {
  (void)in_sizes; (void)n_in; (void)out_size; (void)ws_size;

  const int*   tgt       = (const int*)d_in[0];     // (B,T)
  const int*   len_src   = (const int*)d_in[1];     // (B,)
  // d_in[2] len_tgt unused (fixed T); d_in[5] teacher_forcing == 1 (gold tokens)
  const float* enc_final = (const float*)d_in[3];   // (1,B,H)
  const float* enc_out   = (const float*)d_in[4];   // (S,B,H)
  const float* emb       = (const float*)d_in[6];   // (V,E)
  const float* W_ih      = (const float*)d_in[7];   // (3H, E+H)
  const float* W_hh      = (const float*)d_in[8];   // (3H, H)
  const float* b_ih      = (const float*)d_in[9];
  const float* b_hh      = (const float*)d_in[10];
  const float* W_attn    = (const float*)d_in[11];  // (H,H)
  const float* W_concat  = (const float*)d_in[12];  // (H,2H)
  const float* b_concat  = (const float*)d_in[13];
  const float* W_out     = (const float*)d_in[14];  // (V,H)
  const float* b_out     = (const float*)d_in[15];

  float* out = (float*)d_out;
  float* out_words = out + (long)TSTEPS * B_ * V_;

  // ---- workspace layout ----
  char* ws = (char*)d_ws;
  size_t off = 0;
  auto take = [&](size_t bytes) -> void* {
    void* p = ws + off;
    off = (off + bytes + 255) & ~(size_t)255;
    return p;
  };
  __bf16* wih_bf   = (__bf16*)take((size_t)3 * H_ * (E_ + H_) * 2);
  __bf16* whh_bf   = (__bf16*)take((size_t)3 * H_ * H_ * 2);
  __bf16* wcat_bf  = (__bf16*)take((size_t)H_ * 2 * H_ * 2);
  __bf16* wattn_bf = (__bf16*)take((size_t)H_ * H_ * 2);
  __bf16* wout_bf  = (__bf16*)take((size_t)V_ * H_ * 2);
  __bf16* enc_bf   = (__bf16*)take((size_t)B_ * S_ * H_ * 2);
  float*  enc_proj = (float*)take((size_t)B_ * S_ * H_ * 4);
  float*  h_f      = (float*)take((size_t)B_ * H_ * 4);
  __bf16* h_bf     = (__bf16*)take((size_t)B_ * H_ * 2);
  float*  attnh_f  = (float*)take((size_t)B_ * H_ * 4);
  __bf16* attnh_bf = (__bf16*)take((size_t)B_ * H_ * 2);
  __bf16* x_bf     = (__bf16*)take((size_t)B_ * (E_ + H_) * 2);
  float*  gi       = (float*)take((size_t)B_ * 3 * H_ * 4);
  float*  gh       = (float*)take((size_t)B_ * 3 * H_ * 4);
  __bf16* cat_bf   = (__bf16*)take((size_t)B_ * 2 * H_ * 2);
  float*  logits   = (float*)take((size_t)B_ * V_ * 4);
  float*  stats    = (float*)take((size_t)B_ * 2 * 4);

  const int TPB = 256;

  // ---- one-time conversions (bf16 weights stay resident in L2) ----
  f2bf_kernel<<<cdiv((long)3 * H_ * (E_ + H_), TPB), TPB, 0, stream>>>(W_ih, wih_bf, (long)3 * H_ * (E_ + H_));
  f2bf_kernel<<<cdiv((long)3 * H_ * H_, TPB), TPB, 0, stream>>>(W_hh, whh_bf, (long)3 * H_ * H_);
  f2bf_kernel<<<cdiv((long)H_ * 2 * H_, TPB), TPB, 0, stream>>>(W_concat, wcat_bf, (long)H_ * 2 * H_);
  f2bf_kernel<<<cdiv((long)H_ * H_, TPB), TPB, 0, stream>>>(W_attn, wattn_bf, (long)H_ * H_);
  f2bf_kernel<<<cdiv((long)V_ * H_, TPB), TPB, 0, stream>>>(W_out, wout_bf, (long)V_ * H_);
  enc_transpose_kernel<<<cdiv((long)B_ * S_ * H_, TPB), TPB, 0, stream>>>(enc_out, enc_bf);

  // enc_proj (B*S, H) = enc_bsh @ W_attn^T   (M=4096, N=512, K=512)
  wmma_gemm_kernel<0><<<dim3(cdiv(H_ / 16, 32), (B_ * S_) / 16), TPB, 0, stream>>>(
      enc_bf, wattn_bf, H_, nullptr, enc_proj, H_, nullptr, H_ / 16, H_);

  init_state_kernel<<<cdiv((long)B_ * H_, TPB), TPB, 0, stream>>>(enc_final, h_f, h_bf, attnh_bf);

  // ---- decode loop (serial recurrence over h, attn_h) ----
  for (int t = 0; t < TSTEPS; ++t) {
    build_x_kernel<<<cdiv((long)B_ * (E_ + H_), TPB), TPB, 0, stream>>>(tgt, emb, attnh_bf, x_bf, t);

    // gi = x @ W_ih^T + b_ih    (M=64, N=1536, K=768)
    wmma_gemm_kernel<0><<<dim3(cdiv(3 * H_ / 16, 32), B_ / 16), TPB, 0, stream>>>(
        x_bf, wih_bf, E_ + H_, b_ih, gi, 3 * H_, nullptr, 3 * H_ / 16, E_ + H_);
    // gh = h @ W_hh^T + b_hh    (M=64, N=1536, K=512)
    wmma_gemm_kernel<0><<<dim3(cdiv(3 * H_ / 16, 32), B_ / 16), TPB, 0, stream>>>(
        h_bf, whh_bf, H_, b_hh, gh, 3 * H_, nullptr, 3 * H_ / 16, H_);

    gru_combine_kernel<<<cdiv((long)B_ * H_, TPB), TPB, 0, stream>>>(gi, gh, h_f, h_bf);

    attn_kernel<<<B_, TPB, 0, stream>>>(h_f, h_bf, enc_proj, enc_out, len_src, cat_bf);

    // attn_h = tanh([h|ctx] @ W_concat^T + b)   (M=64, N=512, K=1024)
    wmma_gemm_kernel<1><<<dim3(cdiv(H_ / 16, 32), B_ / 16), TPB, 0, stream>>>(
        cat_bf, wcat_bf, 2 * H_, b_concat, attnh_f, H_, attnh_bf, H_ / 16, 2 * H_);

    // logits = attn_h @ W_out^T + b_out  (M=64, N=32000, K=512) — dominant GEMM
    wmma_gemm_kernel<0><<<dim3(cdiv(V_ / 16, 32), B_ / 16), TPB, 0, stream>>>(
        attnh_bf, wout_bf, H_, b_out, logits, V_, nullptr, V_ / 16, H_);

    logits_stats_kernel<<<B_, TPB, 0, stream>>>(logits, stats, out_words + (long)t * B_);

    write_logp_kernel<<<cdiv((long)B_ * V_, TPB), TPB, 0, stream>>>(
        logits, stats, out + (long)t * B_ * V_);
  }
}
